// WindowAttention_33285996544338
// MI455X (gfx1250) — compile-verified, matrix operand order checked by reading, not run on hardware
//
#include <hip/hip_runtime.h>

typedef __attribute__((ext_vector_type(16))) _Float16 v16h;
typedef __attribute__((ext_vector_type(8)))  float    v8f;

#define WMMA_F16(a, b, c) \
    __builtin_amdgcn_wmma_f32_16x16x32_f16(false, (a), false, (b), (short)0, (c), false, false)

union F16Frag { v16h v; uint4 u[2]; };
union H4Pack  { _Float16 h[4]; uint2 u; };
union H8Pack  { _Float16 h[8]; uint4 u; };

// Load a 16x32 (A) or 32x16 (B) f16 WMMA fragment from a row-major source
// S[row][k], k contiguous (stride in halfs). CDNA5 16-bit operand layout:
// halves 0..7  <- k = kbase + hi*8 .. +7
// halves 8..15 <- k = kbase + 16 + hi*8 .. +7
__device__ inline v16h ld_frag(const _Float16* __restrict__ base, int row, int stride,
                               int kbase, int hi) {
    F16Frag f;
    const _Float16* p = base + row * stride + kbase + hi * 8;
    f.u[0] = *(const uint4*)(p);
    f.u[1] = *(const uint4*)(p + 16);
    return f.v;
}

// ---- prep: f32->f16 weights + combined (rpb bias + shift mask) table ----
// cb layout: [64 win-classes][4 heads][64 m][64 t] f16, padded entries = -20000
__global__ __launch_bounds__(256)
void swin_prep_kernel(const float* __restrict__ qkv_w, const float* __restrict__ proj_w,
                      const float* __restrict__ attn_mask, const float* __restrict__ rpb_table,
                      const int* __restrict__ rpb_index,
                      _Float16* __restrict__ wq, _Float16* __restrict__ wp,
                      _Float16* __restrict__ cb) {
    int i = blockIdx.x * 256 + threadIdx.x;
    if (i < 384 * 128) wq[i] = (_Float16)qkv_w[i];
    else if (i < 384 * 128 + 128 * 128) wp[i - 384 * 128] = (_Float16)proj_w[i - 384 * 128];
    if (i < 64 * 4 * 64 * 64) {
        int n = i & 63, m = (i >> 6) & 63, h = (i >> 12) & 3, w = i >> 14;
        float v = -20000.0f;
        if (m < 49 && n < 49)
            v = rpb_table[rpb_index[m * 49 + n] * 4 + h] + attn_mask[w * 2401 + m * 49 + n];
        cb[i] = (_Float16)v;
    }
}

// ---- fused window attention: 1 block = 1 window, 8 waves ----
__global__ __launch_bounds__(256)
void swin_attn_kernel(const float* __restrict__ x,
                      const _Float16* __restrict__ w_qkv,   // f16 [384][128]
                      const float* __restrict__ qkv_b,      // [384]
                      const _Float16* __restrict__ w_proj,  // f16 [128][128]
                      const float* __restrict__ proj_b,     // [128]
                      const _Float16* __restrict__ cb,      // [64][4][64][64]
                      float* __restrict__ out) {
    extern __shared__ _Float16 smem[];
    _Float16* xs   = smem;                 // [64][128]    8192 halfs
    _Float16* qkv0 = smem + 8192;          // q  [4][64][32] @ +0      (B-source, rows m)
                                           // k  [4][64][32] @ +8192   (A, rows t)
                                           // vT [4][32][64] @ +16384  (A, rows d)
    _Float16* pS   = smem + 32768;         // P row-major [4][64 m][64 t]
    _Float16* ys   = smem + 49152;         // [64][128]

    const int win  = blockIdx.x;
    const int tid  = threadIdx.x;
    const int wave = tid >> 5;
    const int lane = tid & 31;
    const int lo   = lane & 15;
    const int hi   = lane >> 4;

    // ---- stage x tile: float4 loads -> packed 4xf16 b64 stores ----
    const float* xw = x + (size_t)win * 49 * 128;
    for (int i = tid; i < 2048; i += 256) {          // 64 rows * 32 float4
        int m = i >> 5;
        float4 f = {0.f, 0.f, 0.f, 0.f};
        if (m < 49) f = ((const float4*)xw)[i];
        H4Pack p;
        p.h[0] = (_Float16)f.x; p.h[1] = (_Float16)f.y;
        p.h[2] = (_Float16)f.z; p.h[3] = (_Float16)f.w;
        *(uint2*)(xs + i * 4) = p.u;
    }
    __syncthreads();

    // ---- QKV GEMM: [64x128] @ [128x384]; wave -> (mt = wave>>1, 12 ct) ----
    const int mt = wave >> 1;
    const int m0 = mt * 16 + hi * 8;
    {
        v16h axf[4];
        #pragma unroll
        for (int ks = 0; ks < 4; ++ks)
            axf[ks] = ld_frag(xs, mt * 16 + lo, 128, ks * 32, hi);

        if ((wave & 1) == 0) {
            // ct 0..7 -> q, ct 8..11 -> k : strided b16 scatter (row-major [m][d])
            for (int j = 0; j < 12; ++j) {
                int ct = j;
                v8f acc = {};
                #pragma unroll
                for (int ks = 0; ks < 4; ++ks) {
                    v16h b = ld_frag(w_qkv, ct * 16 + lo, 128, ks * 32, hi);
                    acc = WMMA_F16(axf[ks], b, acc);
                }
                float bia = qkv_b[ct * 16 + lo];
                int which = ct >> 3;
                int hh    = (ct >> 1) & 3;
                int d     = ((ct & 1) << 4) + lo;
                int off0  = which * 8192 + hh * 2048 + m0 * 32 + d;
                #pragma unroll
                for (int r = 0; r < 8; ++r)
                    qkv0[off0 + r * 32] = (_Float16)(acc[r] + bia);
            }
        } else {
            // ct 12..15 -> k : strided b16 scatter
            for (int j = 0; j < 4; ++j) {
                int ct = 12 + j;
                v8f acc = {};
                #pragma unroll
                for (int ks = 0; ks < 4; ++ks) {
                    v16h b = ld_frag(w_qkv, ct * 16 + lo, 128, ks * 32, hi);
                    acc = WMMA_F16(axf[ks], b, acc);
                }
                float bia = qkv_b[ct * 16 + lo];
                int hh    = (ct >> 1) & 3;
                int d     = ((ct & 1) << 4) + lo;
                int off0  = 8192 + hh * 2048 + m0 * 32 + d;
                #pragma unroll
                for (int r = 0; r < 8; ++r)
                    qkv0[off0 + r * 32] = (_Float16)(acc[r] + bia);
            }
            // ct 16..23 -> v : transposed store, 8 contiguous halfs = 1 b128
            for (int j = 0; j < 8; ++j) {
                int ct = 16 + j;
                v8f acc = {};
                #pragma unroll
                for (int ks = 0; ks < 4; ++ks) {
                    v16h b = ld_frag(w_qkv, ct * 16 + lo, 128, ks * 32, hi);
                    acc = WMMA_F16(axf[ks], b, acc);
                }
                float bia = qkv_b[ct * 16 + lo];
                int hh    = (ct >> 1) & 3;
                int d     = ((ct & 1) << 4) + lo;
                H8Pack pk;
                #pragma unroll
                for (int r = 0; r < 8; ++r) pk.h[r] = (_Float16)(acc[r] + bia);
                *(uint4*)(qkv0 + 16384 + hh * 2048 + d * 64 + m0) = pk.u;
            }
        }
    }
    __syncthreads();

    // ---- attention, transposed: S^T = K @ Q^T ; wave -> (head, 32 query cols) ----
    const int hd  = wave >> 1;
    const int mtb = (wave & 1) * 2;
    const float scale = 0.17677669529663689f;       // 1/sqrt(32)
    const _Float16* qH  = qkv0 + hd * 2048;         // [64 m][32 d]
    const _Float16* kH  = qkv0 + 8192 + hd * 2048;  // [64 t][32 d]
    const _Float16* vTH = qkv0 + 16384 + hd * 2048; // [32 d][64 t]
    _Float16* pSH = pS + hd * 4096;                 // [64 m][64 t]
    const _Float16* CBw = cb + (((size_t)(win & 63) * 4 + hd) << 12);

    // sc[mi][tt][r]: S^T fragment data; lane owns column m, rows t contiguous in r
    float sc[2][4][8];
    {
        v16h kfr[4], qfr[2];
        #pragma unroll
        for (int tt = 0; tt < 4; ++tt)
            kfr[tt] = ld_frag(kH, tt * 16 + lo, 32, 0, hi);
        #pragma unroll
        for (int mi = 0; mi < 2; ++mi)
            qfr[mi] = ld_frag(qH, (mtb + mi) * 16 + lo, 32, 0, hi);
        #pragma unroll
        for (int mi = 0; mi < 2; ++mi) {
            int m = (mtb + mi) * 16 + lo;
            #pragma unroll
            for (int tt = 0; tt < 4; ++tt) {
                v8f acc = {};
                acc = WMMA_F16(kfr[tt], qfr[mi], acc);
                H8Pack cbv;
                cbv.u = *(const uint4*)(CBw + m * 64 + tt * 16 + hi * 8);
                #pragma unroll
                for (int r = 0; r < 8; ++r)
                    sc[mi][tt][r] = acc[r] * scale + (float)cbv.h[r];
            }
        }
    }

    // ---- softmax over t (register axis + one cross-half shuffle) ----
    #pragma unroll
    for (int mi = 0; mi < 2; ++mi) {
        float mx = -1e30f;
        #pragma unroll
        for (int tt = 0; tt < 4; ++tt)
            #pragma unroll
            for (int r = 0; r < 8; ++r) mx = fmaxf(mx, sc[mi][tt][r]);
        mx = fmaxf(mx, __shfl_xor(mx, 16, 32));
        float sum = 0.f;
        #pragma unroll
        for (int tt = 0; tt < 4; ++tt)
            #pragma unroll
            for (int r = 0; r < 8; ++r) {
                float e = __expf(sc[mi][tt][r] - mx);
                sc[mi][tt][r] = e;
                sum += e;
            }
        sum += __shfl_xor(sum, 16, 32);
        float inv = 1.0f / sum;
        int m = (mtb + mi) * 16 + lo;
        #pragma unroll
        for (int tt = 0; tt < 4; ++tt) {
            H8Pack pk;
            #pragma unroll
            for (int r = 0; r < 8; ++r) pk.h[r] = (_Float16)(sc[mi][tt][r] * inv);
            *(uint4*)(pSH + m * 64 + tt * 16 + hi * 8) = pk.u;   // P row-major
        }
    }
    // P/vT reads below are same-wave or pre-barrier; DS ops are in-order per wave.

    // ---- Y^T = V^T @ P^T : A = vT rows d, B-source = P row-major rows m ----
    {
        v16h vfr[2][2];
        #pragma unroll
        for (int dt = 0; dt < 2; ++dt)
            #pragma unroll
            for (int ks = 0; ks < 2; ++ks)
                vfr[dt][ks] = ld_frag(vTH, dt * 16 + lo, 64, ks * 32, hi);
        #pragma unroll
        for (int mi = 0; mi < 2; ++mi) {
            int m = (mtb + mi) * 16 + lo;
            v16h bfr[2];
            #pragma unroll
            for (int ks = 0; ks < 2; ++ks)
                bfr[ks] = ld_frag(pSH, m, 64, ks * 32, hi);
            #pragma unroll
            for (int dt = 0; dt < 2; ++dt) {
                v8f acc = {};
                #pragma unroll
                for (int ks = 0; ks < 2; ++ks)
                    acc = WMMA_F16(vfr[dt][ks], bfr[ks], acc);
                // lane owns token column m; rows d contiguous -> b128 store
                H8Pack pk;
                #pragma unroll
                for (int r = 0; r < 8; ++r) pk.h[r] = (_Float16)acc[r];
                *(uint4*)(ys + m * 128 + hd * 32 + dt * 16 + hi * 8) = pk.u;
            }
        }
    }
    __syncthreads();

    // ---- output projection: [64x128] @ [128x128] + b; mt fixed per wave ----
    float* ow = out + (size_t)win * 49 * 128;
    {
        v16h ayf[4];
        #pragma unroll
        for (int ks = 0; ks < 4; ++ks)
            ayf[ks] = ld_frag(ys, mt * 16 + lo, 128, ks * 32, hi);
        for (int j = 0; j < 4; ++j) {
            int ct = (wave & 1) * 4 + j;
            v8f acc = {};
            #pragma unroll
            for (int ks = 0; ks < 4; ++ks) {
                v16h b = ld_frag(w_proj, ct * 16 + lo, 128, ks * 32, hi);
                acc = WMMA_F16(ayf[ks], b, acc);
            }
            int c = ct * 16 + lo;
            float pb = proj_b[c];
            #pragma unroll
            for (int r = 0; r < 8; ++r) {
                int m = mt * 16 + hi * 8 + r;
                if (m < 49) ow[m * 128 + c] = acc[r] + pb;
            }
        }
    }
}

extern "C" void kernel_launch(void* const* d_in, const int* in_sizes, int n_in,
                              void* d_out, int out_size, void* d_ws, size_t ws_size,
                              hipStream_t stream) {
    const float* x         = (const float*)d_in[0];
    const float* attn_mask = (const float*)d_in[1];
    const float* qkv_w     = (const float*)d_in[2];
    const float* qkv_b     = (const float*)d_in[3];
    const float* proj_w    = (const float*)d_in[4];
    const float* proj_b    = (const float*)d_in[5];
    const float* rpb_table = (const float*)d_in[6];
    const int*   rpb_index = (const int*)d_in[7];
    float* out = (float*)d_out;

    _Float16* wq = (_Float16*)d_ws;          // 384*128 halfs
    _Float16* wp = wq + 384 * 128;           // 128*128 halfs
    _Float16* cb = wp + 128 * 128;           // 64*4*64*64 halfs (2 MB)

    swin_prep_kernel<<<4096, 256, 0, stream>>>(qkv_w, proj_w, attn_mask,
                                               rpb_table, rpb_index, wq, wp, cb);

    const int smem_bytes = 57344 * (int)sizeof(_Float16);  // 112 KB
    swin_attn_kernel<<<8192, 256, smem_bytes, stream>>>(
        x, wq, qkv_b, wp, proj_b, cb, out);
}
